// Loss_58102317580932
// MI455X (gfx1250) — compile-verified, matrix operand order checked by reading, not run on hardware
//
#include <hip/hip_runtime.h>
#include <hip/hip_bf16.h>
#include <stdint.h>

// CDNA5 / gfx1250: wave32, WMMA (not MFMA).
typedef __attribute__((ext_vector_type(8))) int v8i;

#define K1_ANCH 8                      // anchors per thread in argmax kernel
#define K2_WAVES 8                     // waves per block in focal kernel
#define K2_ITER 4                      // 32-anchor groups per wave
#define K2_ANCH_PER_BLOCK (K2_WAVES * 32 * K2_ITER)   // 1024

// ---------------------------------------------------------------------------
// Kernel 0: zero best-key table and the loss accumulator (both live in d_ws).
// ---------------------------------------------------------------------------
__global__ void k_init(unsigned long long* __restrict__ best,
                       float* __restrict__ acc, int n) {
  int i = blockIdx.x * blockDim.x + threadIdx.x;
  if (i < n) best[i] = 0ull;
  if (i == 0) acc[0] = 0.0f;
}

// ---------------------------------------------------------------------------
// Kernel 1: per (batch, gt) argmax of IoU over all anchors.
// Key = (iou_bits << 32) | ~anchor_idx  -> atomicMax == argmax with
// first-lowest-index tie-break (identical to jnp.argmax).
// ---------------------------------------------------------------------------
__global__ __launch_bounds__(256)
void k_best_anchor(const float* __restrict__ anchors,   // [A,4] xyxy
                   const float* __restrict__ gt,        // [B,N,4] xywh
                   const int*   __restrict__ numobj,    // [B]
                   unsigned long long* __restrict__ best, // [B,N]
                   int A, int N) {
  const int b = blockIdx.y;
  const int num = numobj[b];

  __shared__ float sg[5][64];   // x1,y1,x2,y2,area
  for (int j = threadIdx.x; j < num; j += blockDim.x) {
    float4 g = ((const float4*)gt)[b * N + j];
    sg[0][j] = g.x;        sg[1][j] = g.y;
    sg[2][j] = g.x + g.z;  sg[3][j] = g.y + g.w;
    sg[4][j] = g.z * g.w;
  }
  __syncthreads();

  const int base = blockIdx.x * (256 * K1_ANCH) + threadIdx.x;
  float ax1[K1_ANCH], ay1[K1_ANCH], ax2[K1_ANCH], ay2[K1_ANCH], aar[K1_ANCH];
  int   aidx[K1_ANCH];
#pragma unroll
  for (int i = 0; i < K1_ANCH; ++i) {
    int a  = base + i * 256;
    int ac = a < A ? a : A - 1;        // clamped duplicate never changes argmax
    float4 v = ((const float4*)anchors)[ac];
    ax1[i] = v.x; ay1[i] = v.y; ax2[i] = v.z; ay2[i] = v.w;
    aar[i] = (v.z - v.x) * (v.w - v.y);
    aidx[i] = ac;
  }

  for (int j = 0; j < num; ++j) {      // num is block-uniform
    float gx1 = sg[0][j], gy1 = sg[1][j], gx2 = sg[2][j], gy2 = sg[3][j];
    float ga  = sg[4][j];
    unsigned long long kb = 0ull;
#pragma unroll
    for (int i = 0; i < K1_ANCH; ++i) {
      float lx = fmaxf(ax1[i], gx1), ly = fmaxf(ay1[i], gy1);
      float rx = fminf(ax2[i], gx2), ry = fminf(ay2[i], gy2);
      float w  = fmaxf(rx - lx, 0.0f), h = fmaxf(ry - ly, 0.0f);
      float inter = w * h;
      float iou   = inter / (aar[i] + ga - inter);   // union > 0 always
      unsigned long long key =
          ((unsigned long long)__float_as_uint(iou) << 32) |
          (unsigned)(~(unsigned)aidx[i]);
      kb = key > kb ? key : kb;
    }
    // wave32 max-reduction (two 32-bit shuffles per step)
#pragma unroll
    for (int off = 16; off > 0; off >>= 1) {
      unsigned lo = __shfl_xor((unsigned)(kb & 0xffffffffull), off, 32);
      unsigned hi = __shfl_xor((unsigned)(kb >> 32), off, 32);
      unsigned long long o = ((unsigned long long)hi << 32) | lo;
      kb = o > kb ? o : kb;
    }
    if ((threadIdx.x & 31) == 0)
      atomicMax(&best[b * N + j], kb);
  }
}

// ---------------------------------------------------------------------------
// Kernel 2: recompute match masks (branch-free), reduce counts with
// V_WMMA_I32_16X16X64_IU8 (A = u8 match mask 16x64, B = ones -> every D
// column = per-anchor count), then focal loss with weight (1 + 10*count).
// ---------------------------------------------------------------------------
__global__ __launch_bounds__(256)
void k_focal(const float* __restrict__ thrp,       // [1]
             const float* __restrict__ classes,    // [B,A,2]
             const float* __restrict__ anchors,    // [A,4]
             const float* __restrict__ gt,         // [B,N,4]
             const int*   __restrict__ numobj,     // [B]
             const unsigned long long* __restrict__ best, // [B,N]
             float* __restrict__ acc,
             int A, int N) {
  const int b   = blockIdx.y;
  const int num = numobj[b];
  const float thr = thrp[0];

  __shared__ float sg[5][64];          // x1,y1,x2,y2,area (16B-aligned rows)
  __shared__ int   sbest[64];
  __shared__ int   scnt[K2_WAVES][32];
  __shared__ float sred[K2_WAVES];

  for (int j = threadIdx.x; j < 64; j += blockDim.x) {
    const bool ok = (j < num) & (j < N);
    if (ok) {
      float4 g = ((const float4*)gt)[b * N + j];
      sg[0][j] = g.x;        sg[1][j] = g.y;
      sg[2][j] = g.x + g.z;  sg[3][j] = g.y + g.w;
      sg[4][j] = g.z * g.w;
      sbest[j] = (int)~(unsigned)(best[b * N + j] & 0xffffffffull);
    } else {
      sg[0][j] = 0.f; sg[1][j] = 0.f; sg[2][j] = 0.f; sg[3][j] = 0.f;
      sg[4][j] = 0.f;
      sbest[j] = -1;
    }
  }
  // prefetch this block's class probabilities (global_prefetch_b8)
  {
    const float* cbase = classes +
        2 * ((size_t)b * A + (size_t)blockIdx.x * K2_ANCH_PER_BLOCK);
    __builtin_prefetch(cbase + threadIdx.x * 8, 0, 1);
  }
  __syncthreads();

  const int wave = threadIdx.x >> 5;
  const int lane = threadIdx.x & 31;
  const int m    = lane & 15;          // matrix row owned by this lane
  const int g    = lane >> 4;          // K-half selector (A-matrix layout)

  v8i bones;
#pragma unroll
  for (int i = 0; i < 8; ++i) bones[i] = 0x01010101;   // B = all ones (u8)
  const v8i czero = {0, 0, 0, 0, 0, 0, 0, 0};

  // Per-A-VGPR byte mask for "gt j is valid" (depends only on v,c,g -> hoist).
  unsigned vmask[8];
#pragma unroll
  for (int v = 0; v < 8; ++v) {
    const int jb = ((v >> 1) << 4) + ((v & 1) << 2) + (g << 3);
    unsigned vm = 0u;
#pragma unroll
    for (int c = 0; c < 4; ++c)
      vm |= ((unsigned)((jb + c) < num) * 0xFFu) << (8 * c);
    vmask[v] = vm;
  }

  float lacc = 0.0f;

  for (int it = 0; it < K2_ITER; ++it) {
    const int t0 = blockIdx.x * K2_ANCH_PER_BLOCK + wave * (32 * K2_ITER) + it * 32;
    const int a0 = t0 + m;            // tile-0 anchor for this lane pair
    const int a1 = t0 + 16 + m;       // tile-1 anchor
    const int a0c = a0 < A ? a0 : A - 1;
    const int a1c = a1 < A ? a1 : A - 1;
    const unsigned inA0 = a0 < A ? 0xFFFFFFFFu : 0u;
    const unsigned inA1 = a1 < A ? 0xFFFFFFFFu : 0u;
    const float4 v0 = ((const float4*)anchors)[a0c];
    const float4 v1 = ((const float4*)anchors)[a1c];
    const float ar0 = (v0.z - v0.x) * (v0.w - v0.y);
    const float ar1 = (v1.z - v1.x) * (v1.w - v1.y);

    v8i am0, am1;   // 8-bit A matrices, 16x64, ISA VGPR layout
#pragma unroll
    for (int v = 0; v < 8; ++v) {
      // Lane-half K map: lanes 0-15 hold K = {0-7,16-23,32-39,48-55}, lanes
      // 16-31 the +8 halves; the 4 bytes of VGPR v are consecutive j values,
      // so each GT field row is a single ds_load_b128.
      const int jb = ((v >> 1) << 4) + ((v & 1) << 2) + (g << 3);
      const float4 X1 = *(const float4*)&sg[0][jb];
      const float4 Y1 = *(const float4*)&sg[1][jb];
      const float4 X2 = *(const float4*)&sg[2][jb];
      const float4 Y2 = *(const float4*)&sg[3][jb];
      const float4 GA = *(const float4*)&sg[4][jb];
      const int4   BJ = *(const int4*)&sbest[jb];
      unsigned b0 = 0u, b1 = 0u;
#pragma unroll
      for (int c = 0; c < 4; ++c) {
        const float gx1 = (c == 0) ? X1.x : (c == 1) ? X1.y : (c == 2) ? X1.z : X1.w;
        const float gy1 = (c == 0) ? Y1.x : (c == 1) ? Y1.y : (c == 2) ? Y1.z : Y1.w;
        const float gx2 = (c == 0) ? X2.x : (c == 1) ? X2.y : (c == 2) ? X2.z : X2.w;
        const float gy2 = (c == 0) ? Y2.x : (c == 1) ? Y2.y : (c == 2) ? Y2.z : Y2.w;
        const float ga  = (c == 0) ? GA.x : (c == 1) ? GA.y : (c == 2) ? GA.z : GA.w;
        const int   bj  = (c == 0) ? BJ.x : (c == 1) ? BJ.y : (c == 2) ? BJ.z : BJ.w;
        {
          float lx = fmaxf(v0.x, gx1), ly = fmaxf(v0.y, gy1);
          float rx = fminf(v0.z, gx2), ry = fminf(v0.w, gy2);
          float w  = fmaxf(rx - lx, 0.0f), h = fmaxf(ry - ly, 0.0f);
          float inter = w * h;
          float uni   = ar0 + ga - inter;
          // branch-free: bitwise OR of 0/1 predicates (no short-circuit!)
          unsigned pos = (unsigned)(inter > thr * uni) | (unsigned)(a0 == bj);
          b0 |= pos << (8 * c);
        }
        {
          float lx = fmaxf(v1.x, gx1), ly = fmaxf(v1.y, gy1);
          float rx = fminf(v1.z, gx2), ry = fminf(v1.w, gy2);
          float w  = fmaxf(rx - lx, 0.0f), h = fmaxf(ry - ly, 0.0f);
          float inter = w * h;
          float uni   = ar1 + ga - inter;
          unsigned pos = (unsigned)(inter > thr * uni) | (unsigned)(a1 == bj);
          b1 |= pos << (8 * c);
        }
      }
      am0[v] = (int)(b0 & vmask[v] & inA0);
      am1[v] = (int)(b1 & vmask[v] & inA1);
    }

    // counts[m] = sum_k mask[m,k]  -> matrix pipe, co-executes with VALU
    v8i d0 = __builtin_amdgcn_wmma_i32_16x16x64_iu8(false, am0, false, bones,
                                                    czero, false, false);
    v8i d1 = __builtin_amdgcn_wmma_i32_16x16x64_iu8(false, am1, false, bones,
                                                    czero, false, false);

    // D layout: VGPR r, lanes 0-15 hold M=r, lanes 16-31 hold M=8+r (all N
    // columns equal). Lanes 0 and 16 spill their 16+16 counts to wave-private
    // LDS; DS ops are in-order within a wave, so the readback below is safe.
    if (m == 0) {
#pragma unroll
      for (int r = 0; r < 8; ++r) {
        scnt[wave][g * 8 + r]      = d0[r];
        scnt[wave][16 + g * 8 + r] = d1[r];
      }
    }
    __builtin_amdgcn_wave_barrier();
    const int cnt = scnt[wave][lane];          // anchor t0+lane

    const int a   = t0 + lane;
    const int ac  = a < A ? a : A - 1;
    const float2 cc = ((const float2*)classes)[(size_t)b * A + ac];
    const float p = cnt > 0 ? cc.y : cc.x;     // face prob if positive, else bg
    const float q = 1.0f - p;
    const float f = -(q * q) * logf(p);        // focal, GAMMA = 2
    const float w = 1.0f + 10.0f * (float)cnt; // sum(focal) + 10*sum(cnt*focal)
    lacc += (a < A) ? w * f : 0.0f;
  }

  // wave then block reduction, one f32 atomic per block
#pragma unroll
  for (int off = 16; off > 0; off >>= 1)
    lacc += __shfl_xor(lacc, off, 32);
  if (lane == 0) sred[wave] = lacc;
  __syncthreads();
  if (threadIdx.x == 0) {
    float s = 0.0f;
#pragma unroll
    for (int w = 0; w < K2_WAVES; ++w) s += sred[w];
    atomicAdd(acc, s);
  }
}

// ---------------------------------------------------------------------------
// Kernel 3: finalize -> (total, class_loss, coord_loss=0)
// ---------------------------------------------------------------------------
__global__ void k_final(const float* __restrict__ acc,
                        float* __restrict__ out, float scale) {
  if (threadIdx.x == 0 && blockIdx.x == 0) {
    float c = acc[0] * scale;
    out[0] = c;      // total = class + 0
    out[1] = c;      // class loss
    out[2] = 0.0f;   // coord loss (never accumulated in reference)
  }
}

// ---------------------------------------------------------------------------
extern "C" void kernel_launch(void* const* d_in, const int* in_sizes, int n_in,
                              void* d_out, int out_size, void* d_ws, size_t ws_size,
                              hipStream_t stream) {
  (void)n_in; (void)out_size; (void)ws_size;
  const float* thr     = (const float*)d_in[0];                  // [1]
  const float* classes = (const float*)d_in[1];                  // [B,A,2]
  const float* anchors = (const float*)d_in[2];                  // [A,4]
  const float* gt      = (const float*)d_in[3];                  // [B,N,4]
  const int*   numobj  = (const int*)d_in[4];                    // [B]

  const int B = in_sizes[4];
  const int A = in_sizes[2] / 4;
  const int N = in_sizes[3] / (4 * B);

  unsigned long long* best = (unsigned long long*)d_ws;          // B*N keys
  float* acc = (float*)((char*)d_ws + (size_t)B * N * sizeof(unsigned long long));

  const int initN = B * N;
  k_init<<<(initN + 255) / 256, 256, 0, stream>>>(best, acc, initN);

  dim3 g1((A + 256 * K1_ANCH - 1) / (256 * K1_ANCH), B);
  k_best_anchor<<<g1, 256, 0, stream>>>(anchors, gt, numobj, best, A, N);

  dim3 g2((A + K2_ANCH_PER_BLOCK - 1) / K2_ANCH_PER_BLOCK, B);
  k_focal<<<g2, 256, 0, stream>>>(thr, classes, anchors, gt, numobj, best,
                                  acc, A, N);

  k_final<<<1, 64, 0, stream>>>(acc, (float*)d_out, 0.01f / (float)B);
}
